// ScaledDotProductAttention_88046829568788
// MI455X (gfx1250) — matrix operand order repaired and checked by reading, not compiled, not run on hardware
//
#include <hip/hip_runtime.h>

typedef __attribute__((ext_vector_type(16))) _Float16 v16h;
typedef __attribute__((ext_vector_type(2)))  __fp16   v2fp;
typedef __attribute__((ext_vector_type(8)))  float    v8f;

namespace {
constexpr int S_LEN = 2048;
constexpr int D_DIM = 64;
constexpr int KBLK  = 32;   // keys per block iteration
constexpr int QWG   = 128;  // queries per workgroup (8 waves x 16)
constexpr int KSTR  = 88;   // f16 elems per K_lds row (pad: 176B -> conflict-free b128 reads)
constexpr int VSTR  = 40;   // f16 elems per V^T_lds row (pad: 80B -> conflict-free b128 reads)
}

__device__ __forceinline__ unsigned pk2u(float a, float b) {
    union { v2fp h; unsigned u; } c;
    c.h = __builtin_amdgcn_cvt_pkrtz(a, b);
    return c.u;
}

__global__ void __launch_bounds__(256) fa_fwd(const float* __restrict__ Qg,
                                              const float* __restrict__ Kg,
                                              const float* __restrict__ Vg,
                                              const unsigned char* __restrict__ Mg,
                                              float* __restrict__ Og)
{
    __shared__ _Float16 kls[2][KBLK * KSTR];   // K block, row-major [key][dim], f16
    __shared__ _Float16 vls[2][D_DIM * VSTR];  // V block, transposed [dim][key], f16

    const int tid  = threadIdx.x;
    const int lane = tid & 31;
    const int wave = tid >> 5;
    const int hi   = (lane >> 4) & 1;   // half-wave: 0 = lanes 0-15, 1 = lanes 16-31
    const int lq   = lane & 15;         // this lane's query column within the wave tile

    const int bh = blockIdx.y;
    const int q  = blockIdx.x * QWG + wave * 16 + lq;

    const float* Qb = Qg + (size_t)bh * S_LEN * D_DIM;
    const float* Kb = Kg + (size_t)bh * S_LEN * D_DIM;
    const float* Vb = Vg + (size_t)bh * S_LEN * D_DIM;
    const unsigned char* Mb = Mg + (size_t)bh * S_LEN * S_LEN + (size_t)q * S_LEN;
    float* Ob = Og + (size_t)bh * S_LEN * D_DIM;

    // ---- Q^T as WMMA B-fragments (f32 -> f16). Lane holds dims {8h..8h+7},{16+8h..16+8h+7} of each 32-chunk. ----
    // B 32x16 layout: VGPR v<4: rows {8h+2v,8h+2v+1}; v>=4: rows {8+8h+2v,9+8h+2v} (h = lane/16).
    v16h qf[2];
    {
        const float* qrow = Qb + (size_t)q * D_DIM;
        #pragma unroll
        for (int f = 0; f < 2; ++f) {
            const float* p = qrow + f * 32 + hi * 8;
            float4 a = *(const float4*)(p + 0);
            float4 b = *(const float4*)(p + 4);
            float4 c = *(const float4*)(p + 16);
            float4 d = *(const float4*)(p + 20);
            union { v16h h; unsigned u[8]; } u;
            u.u[0] = pk2u(a.x, a.y); u.u[1] = pk2u(a.z, a.w);
            u.u[2] = pk2u(b.x, b.y); u.u[3] = pk2u(b.z, b.w);
            u.u[4] = pk2u(c.x, c.y); u.u[5] = pk2u(c.z, c.w);
            u.u[6] = pk2u(d.x, d.y); u.u[7] = pk2u(d.z, d.w);
            qf[f] = u.h;
        }
    }

    v8f oacc[4] = {};                  // O^T accum: 4 d-tiles x 8 rows, col = q
    float mrun = -__builtin_inff();    // running max (per query = per lane)
    float lrun = 0.0f;                 // running sum
    const float scale = 0.125f;        // 1/sqrt(64)

    const int skey = tid >> 3;         // staging: key row 0..31
    const int sdim = (tid & 7) * 8;    // staging: dim base 0..56

    // ---- software pipeline: global-load registers for the *next* block ----
    float4 ka, kb4, va, vb4;
    {
        const float* ks = Kb + (size_t)skey * D_DIM + sdim;
        const float* vs = Vb + (size_t)skey * D_DIM + sdim;
        ka  = *(const float4*)(ks);
        kb4 = *(const float4*)(ks + 4);
        va  = *(const float4*)(vs);
        vb4 = *(const float4*)(vs + 4);
    }

    int buf = 0;
    for (int kb = 0; kb < S_LEN; kb += KBLK) {
        // ---- store prefetched block into LDS buffer `buf` ----
        {
            union { unsigned u[4]; float4 f; } w;
            w.u[0] = pk2u(ka.x, ka.y);  w.u[1] = pk2u(ka.z, ka.w);
            w.u[2] = pk2u(kb4.x, kb4.y); w.u[3] = pk2u(kb4.z, kb4.w);
            *(float4*)(&kls[buf][skey * KSTR + sdim]) = w.f;

            _Float16* dst = &vls[buf][sdim * VSTR + skey];
            dst[0 * VSTR] = (_Float16)va.x;  dst[1 * VSTR] = (_Float16)va.y;
            dst[2 * VSTR] = (_Float16)va.z;  dst[3 * VSTR] = (_Float16)va.w;
            dst[4 * VSTR] = (_Float16)vb4.x; dst[5 * VSTR] = (_Float16)vb4.y;
            dst[6 * VSTR] = (_Float16)vb4.z; dst[7 * VSTR] = (_Float16)vb4.w;
        }
        __syncthreads();   // single barrier/iter: double buffer makes this sufficient

        // ---- issue next block's global loads (latency hidden under compute) ----
        {
            const int nkb = (kb + KBLK) & (S_LEN - 1);   // wrap: avoids a branch
            const float* ks = Kb + (size_t)(nkb + skey) * D_DIM + sdim;
            const float* vs = Vb + (size_t)(nkb + skey) * D_DIM + sdim;
            ka  = *(const float4*)(ks);
            kb4 = *(const float4*)(ks + 4);
            va  = *(const float4*)(vs);
            vb4 = *(const float4*)(vs + 4);
        }
        __builtin_prefetch(Mb + kb + KBLK, 0, 3);   // next mask block -> global_prefetch_b8

        // ---- scores S^T = K_blk · Q^T  (two 16x16 key tiles, K-dim = 64 = 2 frags) ----
        v8f s0 = {}, s1 = {};
        #pragma unroll
        for (int f = 0; f < 2; ++f) {
            const int cb = f * 32 + hi * 8;   // A 16x32 f16: chunks {cb..cb+7},{cb+16..cb+23}
            {
                const _Float16* row = &kls[buf][(0 * 16 + lq) * KSTR];
                union { v16h h; float4 f4[2]; } u;
                u.f4[0] = *(const float4*)(row + cb);
                u.f4[1] = *(const float4*)(row + cb + 16);
                s0 = __builtin_amdgcn_wmma_f32_16x16x32_f16(false, u.h, false, qf[f],
                                                            (short)0, s0, false, false);
            }
            {
                const _Float16* row = &kls[buf][(1 * 16 + lq) * KSTR];
                union { v16h h; float4 f4[2]; } u;
                u.f4[0] = *(const float4*)(row + cb);
                u.f4[1] = *(const float4*)(row + cb + 16);
                s1 = __builtin_amdgcn_wmma_f32_16x16x32_f16(false, u.h, false, qf[f],
                                                            (short)0, s1, false, false);
            }
        }

        // ---- scale + mask (byte-bool, True = masked) ----
        uint2 mm0 = *(const uint2*)(Mb + kb + hi * 8);        // keys kb + hi*8 + 0..7
        uint2 mm1 = *(const uint2*)(Mb + kb + 16 + hi * 8);   // keys kb+16 + hi*8 + 0..7
        float sa[8], sb[8];
        #pragma unroll
        for (int r = 0; r < 8; ++r) {
            unsigned b0 = ((r < 4 ? (mm0.x >> (8 * r)) : (mm0.y >> (8 * (r - 4)))) & 0xffu);
            unsigned b1 = ((r < 4 ? (mm1.x >> (8 * r)) : (mm1.y >> (8 * (r - 4)))) & 0xffu);
            sa[r] = b0 ? -1e9f : s0[r] * scale;
            sb[r] = b1 ? -1e9f : s1[r] * scale;
        }

        // ---- online softmax: per-query max/sum (in-lane + one xor-16 shuffle) ----
        float bm = -__builtin_inff();
        #pragma unroll
        for (int r = 0; r < 8; ++r) bm = fmaxf(bm, fmaxf(sa[r], sb[r]));
        bm = fmaxf(bm, __shfl_xor(bm, 16, 32));
        float mnew  = fmaxf(mrun, bm);
        float alpha = __expf(mrun - mnew);

        float ps = 0.0f;
        unsigned p0[4], p1[4];   // f16 pairs: rows {2j,2j+1} of each key tile
        #pragma unroll
        for (int j = 0; j < 4; ++j) {
            float a0 = __expf(sa[2 * j] - mnew);
            float a1 = __expf(sa[2 * j + 1] - mnew);
            float b0 = __expf(sb[2 * j] - mnew);
            float b1 = __expf(sb[2 * j + 1] - mnew);
            ps += (a0 + a1) + (b0 + b1);
            p0[j] = pk2u(a0, a1);
            p1[j] = pk2u(b0, b1);
        }
        ps += __shfl_xor(ps, 16, 32);
        lrun = lrun * alpha + ps;
        mrun = mnew;

        // ---- P^T (32x16) B-fragment from C-layout scores: one xor-16 swap ----
        union { v16h h; unsigned u[8]; } pf;
        #pragma unroll
        for (int j = 0; j < 4; ++j) {
            unsigned x0 = (unsigned)__shfl_xor((int)p0[j], 16, 32);
            unsigned x1 = (unsigned)__shfl_xor((int)p1[j], 16, 32);
            pf.u[j]     = hi ? x1 : p0[j];   // rows {2j,2j+1} (lo) / {16+2j,..} (hi)
            pf.u[4 + j] = hi ? p1[j] : x0;   // rows {8+2j,..} (lo) / {24+2j,..} (hi)
        }

        // ---- O^T += V^T · P^T  (4 d-tiles, single K=32 WMMA each) ----
        #pragma unroll
        for (int dt = 0; dt < 4; ++dt) {
            const _Float16* row = &vls[buf][(dt * 16 + lq) * VSTR];
            union { v16h h; float4 f4[2]; } u;
            u.f4[0] = *(const float4*)(row + hi * 16);
            u.f4[1] = *(const float4*)(row + hi * 16 + 8);
            v8f c = oacc[dt] * alpha;
            oacc[dt] = __builtin_amdgcn_wmma_f32_16x16x32_f16(false, u.h, false, pf.h,
                                                              (short)0, c, false, false);
        }

        buf ^= 1;
    }

    // ---- epilogue: O / l, contiguous stores (C rows -> consecutive d) ----
    float inv = 1.0f / lrun;
    float* orow = Ob + (size_t)q * D_DIM + hi * 8;
    #pragma unroll
    for (int dt = 0; dt < 4; ++dt) {
        v8f o = oacc[dt] * inv;
        float4 w0 = { o[0], o[1], o[2], o[3] };
        float4 w1 = { o[4], o[5], o[6], o[7] };
        *(float4*)(orow + dt * 16)     = w0;
        *(float4*)(orow + dt * 16 + 4) = w1;
    }
}

extern "C" void kernel_launch(void* const* d_in, const int* in_sizes, int n_in,
                              void* d_out, int out_size, void* d_ws, size_t ws_size,
                              hipStream_t stream) {
    (void)in_sizes; (void)n_in; (void)out_size; (void)d_ws; (void)ws_size;
    const float* Q = (const float*)d_in[0];
    const float* K = (const float*)d_in[1];
    const float* V = (const float*)d_in[2];
    const unsigned char* M = (const unsigned char*)d_in[3];
    float* O = (float*)d_out;
    dim3 grid(S_LEN / QWG, 2 * 16);   // (q-blocks, B*H)
    fa_fwd<<<grid, dim3(256), 0, stream>>>(Q, K, V, M, O);
}
